// AttentionHead_41317585387727
// MI455X (gfx1250) — compile-verified, matrix operand order checked by reading, not run on hardware
//
#include <hip/hip_runtime.h>

#define B_   8
#define S_   2048
#define D_   2048
#define H_   128
#define NEG_ (-1.0e9f)

typedef __attribute__((ext_vector_type(16))) _Float16 v16h;
typedef __attribute__((ext_vector_type(8)))  _Float16 v8h;
typedef __attribute__((ext_vector_type(8)))  float    v8f;

// ---------------------------------------------------------------------------
// Fragment helpers (wave32, v_wmma_f32_16x16x32_f16).
// A fragment (16Mx32K f16): lane = m + 16*half;
//   elems 0..7  -> K = half*8 + 0..7 ; elems 8..15 -> K = 16 + half*8 + 0..7
// B fragment (32Kx16N) from an N-major buffer uses the same per-lane pattern.
// rowk0 must be 16-byte aligned.
// ---------------------------------------------------------------------------
__device__ __forceinline__ v16h load_frag(const _Float16* rowk0, int half) {
  v8h lo = *reinterpret_cast<const v8h*>(rowk0 + half * 8);
  v8h hi = *reinterpret_cast<const v8h*>(rowk0 + 16 + half * 8);
  v16h r;
#pragma unroll
  for (int i = 0; i < 8; ++i) { r[i] = lo[i]; r[i + 8] = hi[i]; }
  return r;
}

__device__ __forceinline__ v8f wmma_f16(v16h a, v16h b, v8f c) {
  return __builtin_amdgcn_wmma_f32_16x16x32_f16(false, a, false, b,
                                                (short)0, c, false, false);
}

// Async byte-copy global -> LDS (GLOBAL_LOAD_ASYNC_TO_LDS_B128, GVS mode).
__device__ __forceinline__ void async_copy_b128(unsigned lds_addr,
                                                unsigned gbyte_off,
                                                unsigned long long base) {
  asm volatile("global_load_async_to_lds_b128 %0, %1, %2"
               :: "v"(lds_addr), "v"(gbyte_off), "s"(base) : "memory");
}
__device__ __forceinline__ void async_wait0() {
  asm volatile("s_wait_asynccnt 0" ::: "memory");
}
__device__ __forceinline__ unsigned lds_off(const void* p) {
  return (unsigned)(unsigned long long)p;   // LDS aperture: low 32 bits = offset
}

__device__ __forceinline__ float red_max16(float v) {
#pragma unroll
  for (int off = 8; off; off >>= 1) v = fmaxf(v, __shfl_xor(v, off, 16));
  return v;
}
__device__ __forceinline__ float red_sum16(float v) {
#pragma unroll
  for (int off = 8; off; off >>= 1) v += __shfl_xor(v, off, 16);
  return v;
}

// ---------------------------------------------------------------------------
// Kernel 0: one-time weight transpose+convert: Wt[which][n][k] = f16(W[k][n])
// ---------------------------------------------------------------------------
__global__ __launch_bounds__(256)
void wt_kernel(const float* __restrict__ Wq, const float* __restrict__ Wk,
               const float* __restrict__ Wv, _Float16* __restrict__ wt) {
  const int which = blockIdx.y;
  const float* W = (which == 0) ? Wq : (which == 1) ? Wk : Wv;
  _Float16* dst = wt + (size_t)which * H_ * D_;
  int idx = blockIdx.x * 256 + threadIdx.x;
  int k = idx >> 7, n = idx & 127;
  dst[(size_t)n * D_ + k] = (_Float16)W[(size_t)k * H_ + n];
}

// ---------------------------------------------------------------------------
// Kernel 1: QKV projection, double-buffered.
// Block 256 threads (8 waves), tile 128x128, K-steps of 32, one barrier/step.
// B-tile: async-to-LDS copy of pre-transposed f16 Wt (prefetched one step
// ahead under the WMMAs). A-tile: float4 load -> packed cvt -> b128 store.
// ---------------------------------------------------------------------------
__global__ __launch_bounds__(256)
void qkv_proj_kernel(const float* __restrict__ hidden,
                     const _Float16* __restrict__ wt,
                     const float* __restrict__ bq, const float* __restrict__ bk,
                     const float* __restrict__ bv,
                     _Float16* __restrict__ qh, _Float16* __restrict__ kh,
                     _Float16* __restrict__ vt) {
  __shared__ __align__(16) _Float16 As[2][128][48];   // [m][k], 96B stride
  __shared__ __align__(16) _Float16 Bs[2][128][48];   // [n][k], 96B stride

  const int which = blockIdx.y;
  const float* bias = (which == 0) ? bq : (which == 1) ? bk : bv;
  const unsigned long long wbase =
      (unsigned long long)(wt + (size_t)which * H_ * D_);

  const int tid  = threadIdx.x;
  const int wave = tid >> 5;
  const int lane = tid & 31;
  const int m16  = lane & 15;
  const int half = lane >> 4;
  const int row0 = blockIdx.x * 128;
  const int ar = tid >> 1, ac = (tid & 1) * 16;       // staging coords

  v8f acc[8];
#pragma unroll
  for (int t = 0; t < 8; ++t)
#pragma unroll
    for (int i = 0; i < 8; ++i) acc[t][i] = 0.0f;

  const float4* arow = reinterpret_cast<const float4*>(
      hidden + (size_t)(row0 + ar) * D_ + ac);        // +kk/4 per step

  // --- staging helpers ---
  auto stageB = [&](int kk, int buf) {
#pragma unroll
    for (int i = 0; i < 2; ++i) {
      int c = tid + i * 256;                          // 512 b128 chunks
      int n = c >> 2, seg = c & 3;
      async_copy_b128(lds_off(&Bs[buf][n][seg * 8]),
                      (unsigned)((n * D_ + kk + seg * 8) * 2), wbase);
    }
  };
  auto storeA = [&](const float4* f, int buf) {
    v8h h0, h1;
    h0[0]=(_Float16)f[0].x; h0[1]=(_Float16)f[0].y; h0[2]=(_Float16)f[0].z; h0[3]=(_Float16)f[0].w;
    h0[4]=(_Float16)f[1].x; h0[5]=(_Float16)f[1].y; h0[6]=(_Float16)f[1].z; h0[7]=(_Float16)f[1].w;
    h1[0]=(_Float16)f[2].x; h1[1]=(_Float16)f[2].y; h1[2]=(_Float16)f[2].z; h1[3]=(_Float16)f[2].w;
    h1[4]=(_Float16)f[3].x; h1[5]=(_Float16)f[3].y; h1[6]=(_Float16)f[3].z; h1[7]=(_Float16)f[3].w;
    *reinterpret_cast<v8h*>(&As[buf][ar][ac])     = h0;
    *reinterpret_cast<v8h*>(&As[buf][ar][ac + 8]) = h1;
  };

  // --- prologue: stage step 0 into buffer 0 ---
  stageB(0, 0);
  { float4 f[4] = {arow[0], arow[1], arow[2], arow[3]}; storeA(f, 0); }

  const int NSTEP = D_ / 32;                          // 64
  for (int j = 0; j < NSTEP; ++j) {
    async_wait0();                // our async copies for step j are done
    __syncthreads();              // everyone's stage(j) done, buf[nxt] free
    const int cur = j & 1, nxt = cur ^ 1;
    const bool more = (j + 1) < NSTEP;

    float4 f[4];
    if (more) {                   // prefetch next A tile (global, in shadow)
      const float4* src = arow + (size_t)(j + 1) * 8; // 32 floats = 8 float4
      f[0] = src[0]; f[1] = src[1]; f[2] = src[2]; f[3] = src[3];
      stageB((j + 1) * 32, nxt);  // async DMA next B tile under the WMMAs
    }

    // preload all B fragments, then back-to-back WMMAs
    v16h a = load_frag(&As[cur][wave * 16 + m16][0], half);
    v16h bfr[8];
#pragma unroll
    for (int t = 0; t < 8; ++t)
      bfr[t] = load_frag(&Bs[cur][t * 16 + m16][0], half);
#pragma unroll
    for (int t = 0; t < 8; ++t) acc[t] = wmma_f16(a, bfr[t], acc[t]);

    if (more) storeA(f, nxt);     // convert + b128 stores into next buffer
  }

  const float qscale = (which == 0) ? 0.08838834764831845f : 1.0f;
#pragma unroll
  for (int t = 0; t < 8; ++t) {
    int col = t * 16 + m16;
    float bs = bias[col];
#pragma unroll
    for (int r = 0; r < 8; ++r) {
      int rowg = row0 + wave * 16 + half * 8 + r;
      float v = (acc[t][r] + bs) * qscale;
      if (which == 0) {
        qh[(size_t)rowg * H_ + col] = (_Float16)v;
      } else if (which == 1) {
        kh[(size_t)rowg * H_ + col] = (_Float16)v;
      } else {
        int b = rowg / S_, s = rowg % S_;
        vt[((size_t)b * H_ + col) * S_ + s] = (_Float16)v;   // V^T [b][h][s]
      }
    }
  }
}

// ---------------------------------------------------------------------------
// Kernel 2: causal flash attention, double-buffered K/V staging.
// Grid (S/64, B); 4 waves/block; wave owns 16 q-rows, Q in registers.
// One barrier per key-block; next K/V tiles DMA'd under current WMMAs.
// ---------------------------------------------------------------------------
__global__ __launch_bounds__(128)
void flash_attn_kernel(const _Float16* __restrict__ qh,
                       const _Float16* __restrict__ kh,
                       const _Float16* __restrict__ vt,
                       float* __restrict__ out) {
  __shared__ __align__(16) _Float16 Ks[2][32][144];    // [s][h], 288B stride
  __shared__ __align__(16) _Float16 Vts[2][128][48];   // [h][s], 96B stride
  __shared__ __align__(16) _Float16 Ps[4][16][32];     // per-wave P slab

  const int tid  = threadIdx.x;
  const int wave = tid >> 5;
  const int lane = tid & 31;
  const int m16  = lane & 15;
  const int half = lane >> 4;
  const int b    = blockIdx.y;
  const int qrow0 = blockIdx.x * 64 + wave * 16;

  const _Float16* Qb = qh + ((size_t)b * S_ + qrow0) * H_;
  const unsigned long long kbase = (unsigned long long)(kh + (size_t)b * S_ * H_);
  const unsigned long long vbase = (unsigned long long)(vt + (size_t)b * H_ * S_);

  auto stageKV = [&](int kcol0, int buf) {
#pragma unroll
    for (int i = 0; i < 4; ++i) {                      // K: 512 b128 chunks
      int c = tid + i * 128;
      int r = c >> 4, seg = c & 15;
      async_copy_b128(lds_off(&Ks[buf][r][seg * 8]),
                      (unsigned)((((kcol0 + r) * H_) + seg * 8) * 2), kbase);
    }
#pragma unroll
    for (int i = 0; i < 4; ++i) {                      // V^T: 512 chunks
      int c = tid + i * 128;
      int h = c >> 2, seg = c & 3;
      async_copy_b128(lds_off(&Vts[buf][h][seg * 8]),
                      (unsigned)(((h * S_) + kcol0 + seg * 8) * 2), vbase);
    }
  };

  v16h qa[4];
#pragma unroll
  for (int c = 0; c < 4; ++c)
    qa[c] = load_frag(Qb + (size_t)m16 * H_ + 32 * c, half);

  v8f o[8];
  float m_r[8], l_r[8];
#pragma unroll
  for (int t = 0; t < 8; ++t)
#pragma unroll
    for (int i = 0; i < 8; ++i) o[t][i] = 0.0f;
#pragma unroll
  for (int r = 0; r < 8; ++r) { m_r[r] = -3.0e38f; l_r[r] = 0.0f; }

  const int nj_w = (qrow0 + 15) / 32 + 1;              // this wave's bound
  const int nj_b = (blockIdx.x * 64 + 63) / 32 + 1;    // block-wide bound

  stageKV(0, 0);
  for (int j = 0; j < nj_b; ++j) {
    async_wait0();                 // our stage(j) complete
    __syncthreads();               // all stage(j) visible; buf[nxt] reusable
    const int cur = j & 1;
    if (j + 1 < nj_b) stageKV((j + 1) * 32, cur ^ 1);  // DMA under compute

    if (j < nj_w) {
      const int kcol0 = j * 32;
      // ---- scores 16x32 = Q(16x128) . K^T : preload frags, chain WMMAs ----
      v8f sc[2];
#pragma unroll
      for (int t = 0; t < 2; ++t)
#pragma unroll
        for (int i = 0; i < 8; ++i) sc[t][i] = 0.0f;
      v16h kf[2][4];
#pragma unroll
      for (int t = 0; t < 2; ++t)
#pragma unroll
        for (int c = 0; c < 4; ++c)
          kf[t][c] = load_frag(&Ks[cur][t * 16 + m16][32 * c], half);
#pragma unroll
      for (int c = 0; c < 4; ++c)
#pragma unroll
        for (int t = 0; t < 2; ++t)
          sc[t] = wmma_f16(qa[c], kf[t][c], sc[t]);

      // ---- causal mask + online softmax ----
      float p[2][8];
#pragma unroll
      for (int r = 0; r < 8; ++r) {
        int rowg = qrow0 + half * 8 + r;
#pragma unroll
        for (int t = 0; t < 2; ++t) {
          int col = kcol0 + t * 16 + m16;
          sc[t][r] += (col > rowg) ? NEG_ : 0.0f;
        }
        float bmax = red_max16(fmaxf(sc[0][r], sc[1][r]));
        float mnew = fmaxf(m_r[r], bmax);
        float alpha = __expf(m_r[r] - mnew);
        p[0][r] = __expf(sc[0][r] - mnew);
        p[1][r] = __expf(sc[1][r] - mnew);
        float psum = red_sum16(p[0][r] + p[1][r]);
        l_r[r] = l_r[r] * alpha + psum;
        m_r[r] = mnew;
#pragma unroll
        for (int t = 0; t < 8; ++t) o[t][r] *= alpha;
      }

      // ---- P -> f16 A-fragment via per-wave LDS transpose ----
#pragma unroll
      for (int r = 0; r < 8; ++r)
#pragma unroll
        for (int t = 0; t < 2; ++t)
          Ps[wave][half * 8 + r][t * 16 + m16] = (_Float16)p[t][r];
      asm volatile("s_wait_dscnt 0" ::: "memory");
      v16h pa = load_frag(&Ps[wave][m16][0], half);

      // ---- O += P(16x32) . V(32x128) : preload frags, chain WMMAs ----
      v16h vf[8];
#pragma unroll
      for (int t = 0; t < 8; ++t)
        vf[t] = load_frag(&Vts[cur][t * 16 + m16][0], half);
#pragma unroll
      for (int t = 0; t < 8; ++t) o[t] = wmma_f16(pa, vf[t], o[t]);
    }
  }

  // ---- normalize, write fp32 [b][s][h] ----
#pragma unroll
  for (int r = 0; r < 8; ++r) {
    float inv = 1.0f / l_r[r];
    int rowg = qrow0 + half * 8 + r;
    float* orow = out + ((size_t)b * S_ + rowg) * H_;
#pragma unroll
    for (int t = 0; t < 8; ++t) orow[t * 16 + m16] = o[t][r] * inv;
  }
}

// ---------------------------------------------------------------------------
extern "C" void kernel_launch(void* const* d_in, const int* in_sizes, int n_in,
                              void* d_out, int out_size, void* d_ws, size_t ws_size,
                              hipStream_t stream) {
  const float* hidden = (const float*)d_in[0];
  const float* Wq = (const float*)d_in[1];
  const float* bq = (const float*)d_in[2];
  const float* Wk = (const float*)d_in[3];
  const float* bk = (const float*)d_in[4];
  const float* Wv = (const float*)d_in[5];
  const float* bv = (const float*)d_in[6];
  float* out = (float*)d_out;

  const size_t NSH = (size_t)B_ * S_ * H_;
  _Float16* qh = (_Float16*)d_ws;
  _Float16* kh = qh + NSH;
  _Float16* vt = kh + NSH;
  _Float16* wt = vt + NSH;                 // 3 x [128][2048] f16 (1.5 MB)

  wt_kernel<<<dim3((H_ * D_) / 256, 3), 256, 0, stream>>>(Wq, Wk, Wv, wt);
  qkv_proj_kernel<<<dim3((B_ * S_) / 128, 3), 256, 0, stream>>>(
      hidden, wt, bq, bk, bv, qh, kh, vt);
  flash_attn_kernel<<<dim3(S_ / 64, B_), 128, 0, stream>>>(qh, kh, vt, out);
}
